// AVWGCN_32186484916474
// MI455X (gfx1250) — matrix operand order repaired and checked by reading, not compiled, not run on hardware
//
#include <hip/hip_runtime.h>
#include <hip/hip_bf16.h>

#define NN   5000
#define NPAD 5024   // multiple of 32 for the K loop
#define BB   64
#define CIN  64
#define COUT 64
#define DD   10
#define KCH  2

typedef __attribute__((ext_vector_type(16))) __bf16 v16bf;
typedef __attribute__((ext_vector_type(8)))  __bf16 v8bf;
typedef __attribute__((ext_vector_type(8)))  float  v8f;

static __device__ __forceinline__ v8f wmma_bf16(v16bf a, v16bf b, v8f c) {
  // 8 args: (neg_a, A, neg_b, B, c_mod, C, reuse_a, reuse_b)
  return __builtin_amdgcn_wmma_f32_16x16x32_bf16(false, a, false, b, (short)0, c,
                                                 false, false);
}

// ---------------------------------------------------------------------------
// k0: transpose+convert x (B,N,C) f32 -> xpt (B,C,NPAD) bf16, zero-padded in m
//     one block per (64-row m-tile, batch); LDS tile transpose
// ---------------------------------------------------------------------------
__global__ void k0_transpose(const float* __restrict__ x, __bf16* __restrict__ xpt) {
  __shared__ __attribute__((aligned(32))) __bf16 Tls[64 * 72]; // [c][m], stride 72
  const int t  = threadIdx.x;
  const int m0 = blockIdx.x * 64;
  const int b  = blockIdx.y;
  // load 64(m) x 64(c) tile, coalesced float4 reads; transpose into LDS
#pragma unroll
  for (int p = 0; p < 4; ++p) {
    const int ml = p * 16 + (t >> 4);          // 0..63 local m
    const int c  = (t & 15) * 4;
    const int mg = m0 + ml;
    float4 v = make_float4(0.f, 0.f, 0.f, 0.f);
    if (mg < NN) v = *(const float4*)(x + ((long long)b * NN + mg) * CIN + c);
    Tls[(c + 0) * 72 + ml] = (__bf16)v.x;
    Tls[(c + 1) * 72 + ml] = (__bf16)v.y;
    Tls[(c + 2) * 72 + ml] = (__bf16)v.z;
    Tls[(c + 3) * 72 + ml] = (__bf16)v.w;
  }
  __syncthreads();
  // write channel-major, contiguous 32B chunks along m
  const int c  = t >> 2;
  const int ms = (t & 3) * 16;
  if (m0 + ms < NPAD) {
    v16bf o = *(const v16bf*)(&Tls[c * 72 + ms]);
    *(v16bf*)(xpt + ((long long)b * CIN + c) * NPAD + m0 + ms) = o;
  }
}

// ---------------------------------------------------------------------------
// k1: support = softmax(relu(E E^T)) row-wise; bf16 output, padded cols zeroed
// ---------------------------------------------------------------------------
__global__ void k1_support(const float* __restrict__ E, __bf16* __restrict__ S) {
  __shared__ float row[NN];      // 20 KB logits/exp buffer
  __shared__ float red[256];
  __shared__ float evecLs[DD];
  const int t = threadIdx.x;
  const int n = blockIdx.x;
  if (t < DD) evecLs[t] = E[n * DD + t];
  __syncthreads();
  float e[DD];
#pragma unroll
  for (int d = 0; d < DD; ++d) e[d] = evecLs[d];

  float lmax = 0.0f;                                   // relu output >= 0
  for (int m = t; m < NN; m += 256) {
    const float* em = E + m * DD;
    float z = 0.f;
#pragma unroll
    for (int d = 0; d < DD; ++d) z += e[d] * em[d];
    z = fmaxf(z, 0.0f);
    row[m] = z;
    lmax = fmaxf(lmax, z);
  }
  red[t] = lmax; __syncthreads();
  for (int s = 128; s > 0; s >>= 1) {
    if (t < s) red[t] = fmaxf(red[t], red[t + s]);
    __syncthreads();
  }
  const float mx = red[0]; __syncthreads();

  float lsum = 0.f;
  for (int m = t; m < NN; m += 256) {
    float ez = __expf(row[m] - mx);
    row[m] = ez;
    lsum += ez;
  }
  red[t] = lsum; __syncthreads();
  for (int s = 128; s > 0; s >>= 1) {
    if (t < s) red[t] += red[t + s];
    __syncthreads();
  }
  const float inv = 1.0f / red[0];

  __bf16* out = S + (long long)n * NPAD;
  for (int m = t; m < NN; m += 256) out[m] = (__bf16)(row[m] * inv);
  if (t < NPAD - NN) out[NN + t] = (__bf16)0.f;        // zero K padding
}

// ---------------------------------------------------------------------------
// k2: Y[b,n,c] = sum_m S[n,m] * x[b,m,c]   (the 205 GFLOP batched GEMM)
//     block = 128x64 output tile per batch; 8 waves x 4 16x16 D tiles each
// ---------------------------------------------------------------------------
__global__ void k2_spmm(const __bf16* __restrict__ S, const __bf16* __restrict__ xpt,
                        __bf16* __restrict__ Y) {
  __shared__ __attribute__((aligned(32))) __bf16 Als[128 * 32]; // [row][k]
  __shared__ __attribute__((aligned(32))) __bf16 Bls[64 * 32];  // [col][k]
  const int t    = threadIdx.x;
  const int lane = t & 31;
  const int wave = t >> 5;
  const int tileM = blockIdx.x;          // 0..39
  const int b     = blockIdx.y;          // batch

  v8f acc[4] = {};

  // staging: A = 128 rows x 32 k (one v16bf per thread)
  const int arow = t >> 1, aseg = (t & 1) * 16;
  int arow_g = tileM * 128 + arow; if (arow_g >= NN) arow_g = NN - 1;
  const __bf16* aSrc = S + (long long)arow_g * NPAD + aseg;
  // staging: B = 64 cols x 32 k (one v8bf per thread), already channel-major
  const int bcol = t >> 2, bseg = (t & 3) * 8;
  const __bf16* bSrc = xpt + ((long long)b * CIN + bcol) * NPAD + bseg;

  // per-wave fragment addressing (documented 16-bit A/B VGPR layouts)
  const int rA    = wave * 16 + (lane & 15);
  const int kbase = (lane >> 4) << 3;    // A: K 0..7 / 8..15 per half-wave
  const int koff  = (lane >> 4) << 3;    // B: K 0..7 / 8..15 per half-wave
  const int col   = lane & 15;

  for (int k0 = 0; k0 < NPAD; k0 += 32) {
    *(v16bf*)(&Als[arow * 32 + aseg]) = *(const v16bf*)(aSrc + k0);
    *(v8bf*)(&Bls[bcol * 32 + bseg])  = *(const v8bf*)(bSrc + k0);
    if (k0 + 32 < NPAD) {
      __builtin_prefetch(aSrc + k0 + 32, 0, 0);
      __builtin_prefetch(bSrc + k0 + 32, 0, 0);
    }
    __syncthreads();

    union { v16bf v; v8bf h[2]; } af;
    af.h[0] = *(const v8bf*)(&Als[rA * 32 + kbase]);
    af.h[1] = *(const v8bf*)(&Als[rA * 32 + kbase + 16]);
#pragma unroll
    for (int j = 0; j < 4; ++j) {
      union { v16bf v; v8bf h[2]; } bf;
      bf.h[0] = *(const v8bf*)(&Bls[(j * 16 + col) * 32 + koff]);
      bf.h[1] = *(const v8bf*)(&Bls[(j * 16 + col) * 32 + koff + 16]);
      acc[j] = wmma_bf16(af.v, bf.v, acc[j]);
    }
    __syncthreads();
  }

  // D layout: VGPR r, lane l -> M = r + (l>>4)*8, N = l%16
  const int rowBase = tileM * 128 + wave * 16 + ((lane >> 4) << 3);
#pragma unroll
  for (int r = 0; r < 8; ++r) {
    const int node = rowBase + r;
    if (node < NN) {
      __bf16* yp = Y + ((long long)b * NN + node) * CIN;
#pragma unroll
      for (int j = 0; j < 4; ++j) yp[j * 16 + col] = (__bf16)acc[j][r];
    }
  }
}

// ---------------------------------------------------------------------------
// k3: out[b,n,o] = [x[b,n,:] | Y[b,n,:]] (64x128) @ W_n (128x64) + bias_n
//     one block per node; W_n / bias_n built from pools in LDS
// ---------------------------------------------------------------------------
__global__ void k3_out(const float* __restrict__ E, const float* __restrict__ Wpool,
                       const float* __restrict__ Bpool, const float* __restrict__ x,
                       const __bf16* __restrict__ Y, float* __restrict__ out) {
  __shared__ __attribute__((aligned(32))) __bf16 Wls[64 * 128]; // [o][kidx]
  __shared__ __attribute__((aligned(32))) __bf16 Als[64 * 128]; // [batch][kidx]
  __shared__ float biasLs[COUT];
  __shared__ float evecLs[DD];
  const int t = threadIdx.x;
  const int n = blockIdx.x;
  if (t < DD) evecLs[t] = E[n * DD + t];
  __syncthreads();
  float e[DD];
#pragma unroll
  for (int d = 0; d < DD; ++d) e[d] = evecLs[d];

  // W_n: kidx = kcheb*64 + i; stored transposed [o][kidx] for B-frag loads
  {
    const int o = t & 63, kc = t >> 6;   // kc in 0..3, 32 kidx each
#pragma unroll
    for (int kk = 0; kk < 32; ++kk) {
      const int kidx  = kc * 32 + kk;
      const int kcheb = kidx >> 6;
      const int i     = kidx & 63;
      float acc = 0.f;
#pragma unroll
      for (int d = 0; d < DD; ++d)
        acc += e[d] * Wpool[(((d * KCH + kcheb) * CIN + i) * COUT) + o];
      Wls[o * 128 + kidx] = (__bf16)acc;
    }
  }
  if (t < COUT) {
    float acc = 0.f;
#pragma unroll
    for (int d = 0; d < DD; ++d) acc += e[d] * Bpool[d * COUT + t];
    biasLs[t] = acc;
  }
  // stage A = [x (f32, converted inline) | Y (bf16)] per batch (64 x 128)
  {
    const int b = t >> 2, q = t & 3;     // q0,1 -> x halves; q2,3 -> Y halves
    v8bf* dst = (v8bf*)(&Als[b * 128 + q * 32]);
    if (q < 2) {
      const float4* src = (const float4*)(x + ((long long)b * NN + n) * CIN + q * 32);
#pragma unroll
      for (int c8 = 0; c8 < 4; ++c8) {
        float4 f0 = src[c8 * 2], f1 = src[c8 * 2 + 1];
        v8bf o = { (__bf16)f0.x, (__bf16)f0.y, (__bf16)f0.z, (__bf16)f0.w,
                   (__bf16)f1.x, (__bf16)f1.y, (__bf16)f1.z, (__bf16)f1.w };
        dst[c8] = o;
      }
    } else {
      const v8bf* src = (const v8bf*)(Y + ((long long)b * NN + n) * CIN + (q - 2) * 32);
#pragma unroll
      for (int c8 = 0; c8 < 4; ++c8) dst[c8] = src[c8];
    }
  }
  __syncthreads();

  const int lane = t & 31, wave = t >> 5;
  const int mi = wave & 3, nj0 = (wave >> 2) * 2;
  const int rA    = mi * 16 + (lane & 15);
  const int kbase = (lane >> 4) << 3;
  const int koff  = (lane >> 4) << 3;
  const int c0    = nj0 * 16 + (lane & 15);
  const int c1    = c0 + 16;
  v8f acc0 = {}, acc1 = {};
#pragma unroll
  for (int ks = 0; ks < 128; ks += 32) {
    union { v16bf v; v8bf h[2]; } af;
    af.h[0] = *(const v8bf*)(&Als[rA * 128 + ks + kbase]);
    af.h[1] = *(const v8bf*)(&Als[rA * 128 + ks + kbase + 16]);
    union { v16bf v; v8bf h[2]; } b0, b1;
    b0.h[0] = *(const v8bf*)(&Wls[c0 * 128 + ks + koff]);
    b0.h[1] = *(const v8bf*)(&Wls[c0 * 128 + ks + koff + 16]);
    b1.h[0] = *(const v8bf*)(&Wls[c1 * 128 + ks + koff]);
    b1.h[1] = *(const v8bf*)(&Wls[c1 * 128 + ks + koff + 16]);
    acc0 = wmma_bf16(af.v, b0.v, acc0);
    acc1 = wmma_bf16(af.v, b1.v, acc1);
  }

  const int bBase = mi * 16 + ((lane >> 4) << 3);
#pragma unroll
  for (int r = 0; r < 8; ++r) {
    const int b = bBase + r;
    float* op = out + ((long long)b * NN + n) * COUT;
    op[c0] = acc0[r] + biasLs[c0];
    op[c1] = acc1[r] + biasLs[c1];
  }
}

// ---------------------------------------------------------------------------
extern "C" void kernel_launch(void* const* d_in, const int* in_sizes, int n_in,
                              void* d_out, int out_size, void* d_ws, size_t ws_size,
                              hipStream_t stream) {
  const float* x  = (const float*)d_in[0];   // (B,N,CIN)
  const float* E  = (const float*)d_in[1];   // (N,D)
  const float* Wp = (const float*)d_in[2];   // (D,K,CIN,COUT)
  const float* Bp = (const float*)d_in[3];   // (D,COUT)
  float* out = (float*)d_out;                // (B,N,COUT)

  auto align256 = [](size_t v) { return (v + 255) & ~(size_t)255; };
  char* ws = (char*)d_ws;
  size_t off = 0;
  __bf16* S   = (__bf16*)(ws + off); off += align256((size_t)NN * NPAD * 2);       // 50.2 MB
  __bf16* xpt = (__bf16*)(ws + off); off += align256((size_t)BB * CIN * NPAD * 2); // 41.2 MB
  __bf16* Y   = (__bf16*)(ws + off); off += align256((size_t)BB * NN * CIN * 2);   // 41.0 MB
  (void)off; (void)ws_size; (void)in_sizes; (void)n_in; (void)out_size;

  dim3 g0((NN + 63) / 64, BB);               // 79 x 64
  k0_transpose<<<g0, 256, 0, stream>>>(x, xpt);
  k1_support<<<NN, 256, 0, stream>>>(E, S);
  dim3 g2((NN + 127) / 128, BB);             // 40 x 64
  k2_spmm<<<g2, 256, 0, stream>>>(S, xpt, Y);
  k3_out<<<NN, 256, 0, stream>>>(E, Wp, Bp, x, Y, out);
}
